// MultiheadSelfAttention_43267500540574
// MI455X (gfx1250) — matrix operand order repaired and checked by reading, not compiled, not run on hardware
//
#include <hip/hip_runtime.h>
#include <hip/hip_bf16.h>

// ---------------------------------------------------------------------------
// Multi-head causal self-attention for MI455X (gfx1250, wave32, WMMA).
//   B=2, S=2048, D=1024, H=16, d_k=64.  fp32 in/out, bf16 WMMA internally.
// Round 2: DPP ROW_XMASK softmax reductions (no ds_bpermute), double-buffered
//          LDS in the projection GEMMs to overlap global loads with WMMA.
// ---------------------------------------------------------------------------

#define DMODEL 1024
#define SEQ    2048
#define BATCH  2
#define NHEAD  16
#define DK     64
#define MROWS  (BATCH * SEQ)      // 4096

typedef __attribute__((ext_vector_type(16))) __bf16       v16bf;
typedef __attribute__((ext_vector_type(8)))  __bf16       v8bf;
typedef __attribute__((ext_vector_type(4)))  __bf16       v4bf;
typedef __attribute__((ext_vector_type(8)))  float        v8f;
typedef __attribute__((ext_vector_type(4)))  float        v4f;
typedef __attribute__((ext_vector_type(4)))  unsigned int u32x4;

static __device__ __forceinline__ v8f wmma_bf16(v16bf a, v16bf b, v8f c) {
  // D = A(16x32) * B(32x16) + C, fp32 accumulate
  return __builtin_amdgcn_wmma_f32_16x16x32_bf16(
      /*neg_a=*/false, a, /*neg_b=*/false, b,
      /*c_mod=*/(short)0, c, /*reuse_a=*/false, /*reuse_b=*/false);
}

// A-fragment (16x32 bf16): lane L<16 holds row M=L, K = {k0..k0+7, k0+16..k0+23}
// with k0 = 0; lanes 16..31 hold the same rows with k0 = 8.
static __device__ __forceinline__ v16bf load_frag_split(const __bf16* lo,
                                                        const __bf16* hi) {
  union { u32x4 q[2]; v16bf v; } u;
  u.q[0] = *reinterpret_cast<const u32x4*>(lo);
  u.q[1] = *reinterpret_cast<const u32x4*>(hi);
  return u.v;
}

// B-fragment (32x16 bf16), this file's k order (A frags use it too): lane L holds col N=L%16, K k0..k0+15
// (k0 = 0 for lanes 0..15, 16 for lanes 16..31) -> one contiguous 32B run.
static __device__ __forceinline__ v16bf load_frag_contig(const __bf16* p) {
  union { u32x4 q[2]; v16bf v; } u;
  u.q[0] = *reinterpret_cast<const u32x4*>(p);
  u.q[1] = *reinterpret_cast<const u32x4*>(p + 8);
  return u.v;
}

// ---- DPP ROW_XMASK lane-xor within each 16-lane row (VALU, no LDS traffic)
template <int MASK>
static __device__ __forceinline__ float dpp_xor16(float v) {
  return __builtin_bit_cast(
      float, __builtin_amdgcn_update_dpp(0, __builtin_bit_cast(int, v),
                                         0x160 | MASK, 0xf, 0xf, true));
}
static __device__ __forceinline__ float half_max16(float v) {
  v = fmaxf(v, dpp_xor16<1>(v));
  v = fmaxf(v, dpp_xor16<2>(v));
  v = fmaxf(v, dpp_xor16<4>(v));
  v = fmaxf(v, dpp_xor16<8>(v));
  return v;
}
static __device__ __forceinline__ float half_sum16(float v) {
  v += dpp_xor16<1>(v);
  v += dpp_xor16<2>(v);
  v += dpp_xor16<4>(v);
  v += dpp_xor16<8>(v);
  return v;
}

// ---------------------------------------------------------------------------
// Kernel 1: fp32 -> bf16 conversion (X and the four weight matrices)
// ---------------------------------------------------------------------------
__global__ __launch_bounds__(256) void cvt_f32_to_bf16(
    const float* __restrict__ src, __bf16* __restrict__ dst, int n4) {
  int i = blockIdx.x * 256 + threadIdx.x;
  if (i >= n4) return;
  v4f  f = reinterpret_cast<const v4f*>(src)[i];
  v4bf o = __builtin_convertvector(f, v4bf);
  reinterpret_cast<v4bf*>(dst)[i] = o;
}

// ---------------------------------------------------------------------------
// Kernel 2: GEMM  C[m,n] = sum_k A[m,k] * W[n,k]   (A: 4096x1024, W: 1024x1024)
//   MODE 0: store bf16 as [b, h, s, dk]        (Q, K)
//   MODE 1: store bf16 transposed [b, h, dk, s] (V -> contiguous B-frags in PV)
//   MODE 2: store fp32 row-major [m, n]         (output projection)
// 128x128 tile / workgroup, 8 waves (2x4), 64x32 per wave, K-step 32.
// Double-buffered LDS: next tile's global loads are issued before the WMMAs,
// written to the alternate buffer after, one barrier per K-step.
// ---------------------------------------------------------------------------
#define LDS_STRIDE 40   // 32 + 8 bf16 pad (80B rows, bank-conflict friendly)

template <int MODE>
__global__ __launch_bounds__(256) void gemm_bf16_wmma(
    const __bf16* __restrict__ A, const __bf16* __restrict__ W, void* dstv) {
  __shared__ __bf16 ldsA[2][128 * LDS_STRIDE];
  __shared__ __bf16 ldsB[2][128 * LDS_STRIDE];

  const int tid  = threadIdx.x;
  const int lane = tid & 31, wave = tid >> 5;
  const int wm = wave >> 2, wn = wave & 3;         // 2 x 4 wave grid
  const int lr = lane & 15, lhi = lane >> 4;
  const int rowBase = blockIdx.y * 128;
  const int colBase = blockIdx.x * 128;

  // each thread owns two 8-bf16 chunks of the 128x32 tiles
  const int c0 = tid, c1 = tid + 256;
  const int r0 = c0 >> 2, g0 = (c0 & 3) * 8;
  const int r1 = c1 >> 2, g1 = (c1 & 3) * 8;
  const __bf16* gA0 = A + (size_t)(rowBase + r0) * DMODEL + g0;
  const __bf16* gB0 = W + (size_t)(colBase + r0) * DMODEL + g0;
  const __bf16* gA1 = A + (size_t)(rowBase + r1) * DMODEL + g1;
  const __bf16* gB1 = W + (size_t)(colBase + r1) * DMODEL + g1;
  const int dA0 = r0 * LDS_STRIDE + g0, dA1 = r1 * LDS_STRIDE + g1;

  v8f acc[4][2];
#pragma unroll
  for (int mi = 0; mi < 4; ++mi)
#pragma unroll
    for (int ni = 0; ni < 2; ++ni)
#pragma unroll
      for (int e = 0; e < 8; ++e) acc[mi][ni][e] = 0.0f;

  const int NT = DMODEL / 32;

  // prologue: stage K-tile 0 into buffer 0
  {
    u32x4 ta0 = *reinterpret_cast<const u32x4*>(gA0);
    u32x4 tb0 = *reinterpret_cast<const u32x4*>(gB0);
    u32x4 ta1 = *reinterpret_cast<const u32x4*>(gA1);
    u32x4 tb1 = *reinterpret_cast<const u32x4*>(gB1);
    *reinterpret_cast<u32x4*>(&ldsA[0][dA0]) = ta0;
    *reinterpret_cast<u32x4*>(&ldsB[0][dA0]) = tb0;
    *reinterpret_cast<u32x4*>(&ldsA[0][dA1]) = ta1;
    *reinterpret_cast<u32x4*>(&ldsB[0][dA1]) = tb1;
  }
  __syncthreads();

  for (int kt = 0; kt < NT; ++kt) {
    const int cur = kt & 1, nxt = cur ^ 1;

    // issue next tile's global loads early (latency hidden under WMMAs)
    u32x4 ta0, tb0, ta1, tb1;
    const bool more = (kt + 1) < NT;
    if (more) {
      const int ko = (kt + 1) * 32;
      ta0 = *reinterpret_cast<const u32x4*>(gA0 + ko);
      tb0 = *reinterpret_cast<const u32x4*>(gB0 + ko);
      ta1 = *reinterpret_cast<const u32x4*>(gA1 + ko);
      tb1 = *reinterpret_cast<const u32x4*>(gB1 + ko);
    }

    // fragments + 8 WMMAs on the current buffer
    v16bf afr[4], bfr[2];
#pragma unroll
    for (int mi = 0; mi < 4; ++mi) {
      const __bf16* p = &ldsA[cur][(wm * 64 + mi * 16 + lr) * LDS_STRIDE + lhi * 16];
      afr[mi] = load_frag_contig(p);
    }
#pragma unroll
    for (int ni = 0; ni < 2; ++ni) {
      const __bf16* p = &ldsB[cur][(wn * 32 + ni * 16 + lr) * LDS_STRIDE + lhi * 16];
      bfr[ni] = load_frag_contig(p);
    }
#pragma unroll
    for (int mi = 0; mi < 4; ++mi)
#pragma unroll
      for (int ni = 0; ni < 2; ++ni)
        acc[mi][ni] = wmma_bf16(afr[mi], bfr[ni], acc[mi][ni]);

    if (more) {
      *reinterpret_cast<u32x4*>(&ldsA[nxt][dA0]) = ta0;
      *reinterpret_cast<u32x4*>(&ldsB[nxt][dA0]) = tb0;
      *reinterpret_cast<u32x4*>(&ldsA[nxt][dA1]) = ta1;
      *reinterpret_cast<u32x4*>(&ldsB[nxt][dA1]) = tb1;
    }
    __syncthreads();
  }

  // ---- epilogue: C-frag element (M,N) lives at vgpr e, lane = N + 16*(M/8)
#pragma unroll
  for (int mi = 0; mi < 4; ++mi) {
#pragma unroll
    for (int ni = 0; ni < 2; ++ni) {
      const int m_base = rowBase + wm * 64 + mi * 16 + 8 * lhi;  // + e
      const int col    = colBase + wn * 32 + ni * 16 + lr;
      if (MODE == 0) {                       // bf16 [b,h,s,dk]
        __bf16* dst = (__bf16*)dstv;
        const int h = col >> 6, kk = col & 63;
#pragma unroll
        for (int e = 0; e < 8; ++e) {
          const int m = m_base + e, b = m >> 11, s = m & (SEQ - 1);
          dst[(((size_t)(b * NHEAD + h) * SEQ + s) * DK) + kk] = (__bf16)acc[mi][ni][e];
        }
      } else if (MODE == 1) {                // bf16 transposed [b,h,dk,s]
        __bf16* dst = (__bf16*)dstv;
        const int b = m_base >> 11, sb = m_base & (SEQ - 1);
        const int h = col >> 6, dv = col & 63;
        v8bf o;
#pragma unroll
        for (int e = 0; e < 8; ++e) o[e] = (__bf16)acc[mi][ni][e];
        *reinterpret_cast<v8bf*>(
            &dst[((size_t)(b * NHEAD + h) * DK + dv) * SEQ + sb]) = o;
      } else {                               // fp32 row-major [m,n]
        float* dst = (float*)dstv;
#pragma unroll
        for (int e = 0; e < 8; ++e)
          dst[(size_t)(m_base + e) * DMODEL + col] = acc[mi][ni][e];
      }
    }
  }
}

// ---------------------------------------------------------------------------
// Kernel 3: causal flash attention.
//   Grid (S/128, B*H), 256 threads = 8 waves; each wave owns 16 query rows,
//   streams 32-key tiles up to its causal bound, online softmax in fp32 with
//   DPP ROW_XMASK reductions (rows live in 16-lane halves).
//   Q,K: bf16 [b,h,s,64];  V: bf16 [b,h,64,s];  out AO: bf16 [b,s,1024].
// ---------------------------------------------------------------------------
__global__ __launch_bounds__(256) void attn_flash_wmma(
    const __bf16* __restrict__ Qh, const __bf16* __restrict__ Kh,
    const __bf16* __restrict__ Vt, __bf16* __restrict__ AO) {
  __shared__ __bf16 ldsP[8][16 * LDS_STRIDE];   // per-wave P scratch (16x32)

  const int tid  = threadIdx.x;
  const int lane = tid & 31, wave = tid >> 5;
  const int lr = lane & 15, lhi = lane >> 4;
  const int bh = blockIdx.y;                    // b*16 + h
  const int b  = bh >> 4, h = bh & 15;
  const int q0 = blockIdx.x * 128 + wave * 16;

  const __bf16* Qp = Qh + (size_t)bh * SEQ * DK;
  const __bf16* Kp = Kh + (size_t)bh * SEQ * DK;
  const __bf16* Vp = Vt + (size_t)bh * DK * SEQ;

  // softmax scale folded with log2(e) so we can use exp2
  const float scl = 0.125f * 1.44269504088896f;

  // Q A-fragments (16 rows x 64 dk = two K-steps), straight from global
  v16bf qf[2];
#pragma unroll
  for (int j = 0; j < 2; ++j) {
    const __bf16* p = Qp + (size_t)(q0 + lr) * DK + j * 32 + lhi * 16;
    qf[j] = load_frag_contig(p);
  }

  float m_i[8], l_i[8];
  v8f   oacc[4];
#pragma unroll
  for (int e = 0; e < 8; ++e) { m_i[e] = -1e30f; l_i[e] = 0.0f; }
#pragma unroll
  for (int nt = 0; nt < 4; ++nt)
#pragma unroll
    for (int e = 0; e < 8; ++e) oacc[nt][e] = 0.0f;

  const int ntiles = (q0 + 15) / 32 + 1;        // causal: keys <= q0+15
  for (int t = 0; t < ntiles; ++t) {
    const int kb = t * 32;

    // ---- scores S = Q * K^T  (16 x 32), 4 WMMAs
    v8f sc[2];
#pragma unroll
    for (int nt = 0; nt < 2; ++nt) {
#pragma unroll
      for (int e = 0; e < 8; ++e) sc[nt][e] = 0.0f;
#pragma unroll
      for (int j = 0; j < 2; ++j) {
        const __bf16* p = Kp + (size_t)(kb + nt * 16 + lr) * DK + j * 32 + lhi * 16;
        sc[nt] = wmma_bf16(qf[j], load_frag_contig(p), sc[nt]);
      }
    }

    // ---- scale + causal mask + online softmax (DPP reductions)
    float fscale[8];
#pragma unroll
    for (int e = 0; e < 8; ++e) {
      const int q = q0 + e + 8 * lhi;
      float v0 = sc[0][e] * scl; if (kb + lr      > q) v0 = -1e30f;
      float v1 = sc[1][e] * scl; if (kb + 16 + lr > q) v1 = -1e30f;
      const float mrow = half_max16(fmaxf(v0, v1));
      const float mn   = fmaxf(m_i[e], mrow);
      const float fe   = exp2f(m_i[e] - mn);
      const float p0   = exp2f(v0 - mn);
      const float p1   = exp2f(v1 - mn);
      l_i[e] = l_i[e] * fe + half_sum16(p0 + p1);
      m_i[e] = mn;
      sc[0][e] = p0; sc[1][e] = p1;
      fscale[e] = fe;
    }
#pragma unroll
    for (int nt = 0; nt < 4; ++nt)
#pragma unroll
      for (int e = 0; e < 8; ++e) oacc[nt][e] *= fscale[e];

    // ---- P: C-layout -> A-layout via per-wave LDS round trip (bf16)
    __bf16* pw = &ldsP[wave][0];
#pragma unroll
    for (int e = 0; e < 8; ++e) {
      const int r = e + 8 * lhi;
      pw[r * LDS_STRIDE + lr]      = (__bf16)sc[0][e];
      pw[r * LDS_STRIDE + 16 + lr] = (__bf16)sc[1][e];
    }
    asm volatile("s_wait_dscnt 0" ::: "memory");   // stores visible to reads
    const __bf16* pp = &pw[lr * LDS_STRIDE + lhi * 16];
    const v16bf pf = load_frag_contig(pp);

    // ---- O += P * V   (V transposed -> contiguous B-frags), 4 WMMAs
#pragma unroll
    for (int nt = 0; nt < 4; ++nt) {
      const __bf16* vp = Vp + (size_t)(nt * 16 + lr) * SEQ + kb + lhi * 16;
      oacc[nt] = wmma_bf16(pf, load_frag_contig(vp), oacc[nt]);
    }
  }

  // ---- normalize and store AO[b, s, h*64 + dv] (bf16)
#pragma unroll
  for (int nt = 0; nt < 4; ++nt)
#pragma unroll
    for (int e = 0; e < 8; ++e) {
      const int q = q0 + e + 8 * lhi;
      AO[((size_t)(b * SEQ + q)) * DMODEL + h * DK + nt * 16 + lr] =
          (__bf16)(oacc[nt][e] * (1.0f / l_i[e]));
    }
}

// ---------------------------------------------------------------------------
// Host-side launch
// ---------------------------------------------------------------------------
extern "C" void kernel_launch(void* const* d_in, const int* in_sizes, int n_in,
                              void* d_out, int out_size, void* d_ws, size_t ws_size,
                              hipStream_t stream) {
  const float* X  = (const float*)d_in[0];
  const float* Wq = (const float*)d_in[1];
  const float* Wk = (const float*)d_in[2];
  const float* Wv = (const float*)d_in[3];
  const float* Wo = (const float*)d_in[4];
  float* out = (float*)d_out;

  char* ws = (char*)d_ws;
  const size_t XB = (size_t)MROWS * DMODEL * sizeof(__bf16);           // 8 MB
  const size_t WB = (size_t)DMODEL * DMODEL * sizeof(__bf16);          // 2 MB
  __bf16* XH  = (__bf16*)(ws);
  __bf16* WQH = (__bf16*)(ws + XB);
  __bf16* WKH = (__bf16*)(ws + XB + WB);
  __bf16* WVH = (__bf16*)(ws + XB + 2 * WB);
  __bf16* WOH = (__bf16*)(ws + XB + 3 * WB);
  __bf16* QH  = (__bf16*)(ws + XB + 4 * WB);
  __bf16* KH  = (__bf16*)(ws + 2 * XB + 4 * WB);
  __bf16* VT  = (__bf16*)(ws + 3 * XB + 4 * WB);
  __bf16* AO  = (__bf16*)(ws + 4 * XB + 4 * WB);

  // 1) fp32 -> bf16
  const int nx4 = MROWS * DMODEL / 4, nw4 = DMODEL * DMODEL / 4;
  cvt_f32_to_bf16<<<(nx4 + 255) / 256, 256, 0, stream>>>(X,  XH,  nx4);
  cvt_f32_to_bf16<<<(nw4 + 255) / 256, 256, 0, stream>>>(Wq, WQH, nw4);
  cvt_f32_to_bf16<<<(nw4 + 255) / 256, 256, 0, stream>>>(Wk, WKH, nw4);
  cvt_f32_to_bf16<<<(nw4 + 255) / 256, 256, 0, stream>>>(Wv, WVH, nw4);
  cvt_f32_to_bf16<<<(nw4 + 255) / 256, 256, 0, stream>>>(Wo, WOH, nw4);

  // 2) Q/K/V projections (V stored transposed for the P*V B-fragments)
  dim3 ggrid(DMODEL / 128, MROWS / 128);   // (8, 32)
  gemm_bf16_wmma<0><<<ggrid, 256, 0, stream>>>(XH, WQH, QH);
  gemm_bf16_wmma<0><<<ggrid, 256, 0, stream>>>(XH, WKH, KH);
  gemm_bf16_wmma<1><<<ggrid, 256, 0, stream>>>(XH, WVH, VT);

  // 3) causal flash attention
  dim3 agrid(SEQ / 128, BATCH * NHEAD);    // (16, 32)
  attn_flash_wmma<<<agrid, 256, 0, stream>>>(QH, KH, VT, AO);

  // 4) output projection -> fp32 d_out
  gemm_bf16_wmma<2><<<ggrid, 256, 0, stream>>>(AO, WOH, out);
}
